// Qwen35MoEDecoderLayer_22797686407846
// MI455X (gfx1250) — matrix operand order repaired and checked by reading, not compiled, not run on hardware
//
#include <hip/hip_runtime.h>
#include <cstdint>
#include <cstddef>

// ---------------------------------------------------------------------------
// Qwen3.5-MoE decoder layer for MI455X (gfx1250).
// All GEMMs: v_wmma_f32_16x16x32_bf16, B operands pre-transposed to [N][K],
// double-buffered LDS fed by GLOBAL_LOAD_ASYNC_TO_LDS_B128 when available.
// ---------------------------------------------------------------------------

typedef __bf16 bf16;
typedef __attribute__((ext_vector_type(8)))  __bf16 bf16x8;
typedef __attribute__((ext_vector_type(16))) __bf16 bf16x16;
typedef __attribute__((ext_vector_type(8)))  float  f32x8;
typedef int v4i_ __attribute__((vector_size(16)));

constexpr int NS  = 2048;   // sequence
constexpr int ND  = 2048;   // hidden dim
constexpr int NH  = 16;     // heads
constexpr int NDH = 128;    // head dim
constexpr int NHD = 2048;   // H*DH
constexpr int NE  = 8;      // experts
constexpr int NF  = 768;    // moe ffn
constexpr int NSF = 4096;   // shared ffn
constexpr float EPSV = 1e-6f;

#if defined(__gfx1250__) && \
    __has_builtin(__builtin_amdgcn_global_load_async_to_lds_b128) && \
    __has_builtin(__builtin_amdgcn_s_wait_asynccnt)
#define USE_ASYNC_LDS 1
#else
#define USE_ASYNC_LDS 0
#endif

#if USE_ASYNC_LDS
__device__ __forceinline__ void async_ld128(const bf16* g, bf16* l)
{
    __builtin_amdgcn_global_load_async_to_lds_b128(
        (__attribute__((address_space(1))) v4i_*)(void*)g,
        (__attribute__((address_space(3))) v4i_*)(void*)l, 0, 0);
}
#endif

// ---------------------------------------------------------------------------
// WMMA bf16 GEMM (always NT):  C[M,N](f32) = A[M,K] x B_t[N,K] (bf16).
// Block tile 128x128, BK=32; 8 waves, wave tile 64x32 = 4x2 fragments.
// EPI: 0 = store, 1 = store + add[idx] (residual),
//      2 = C += rowscale[row*rss]*acc, 3 = C = rowscale[row*rss]*acc.
// All dims multiples of the tile.
// ---------------------------------------------------------------------------
template <int EPI>
__global__ __launch_bounds__(256)
void k_gemm_wmma(const bf16* __restrict__ Ag, const bf16* __restrict__ Bg,
                 float* __restrict__ Cg, const float* __restrict__ Xg,
                 int K, int lda, int ldb, int ldc,
                 long strA, long strB, long strC, int rss)
{
    __shared__ __align__(32) bf16 sA[2][128 * 32];
    __shared__ __align__(32) bf16 sB[2][128 * 32];

    const int z = blockIdx.z;
    Ag += (long)z * strA;
    Bg += (long)z * strB;
    Cg += (long)z * strC;

    const int m0   = blockIdx.y * 128;
    const int n0   = blockIdx.x * 128;
    const int tid  = threadIdx.x;
    const int lane = tid & 31;
    const int w    = tid >> 5;
    const int wm   = w >> 2;                 // 0..1
    const int wn   = w & 3;                  // 0..3
    const int khalf = lane >> 4;
    const int l15   = lane & 15;

    // per-thread tile chunk: rows {row0, row0+64}, 8 bf16 at column cc0*8
    const int row0 = tid >> 2;
    const int cc8  = (tid & 3) << 3;
    const int soff = row0 * 32 + cc8;        // LDS offset of first chunk

    const f32x8 vzero = {0.f, 0.f, 0.f, 0.f, 0.f, 0.f, 0.f, 0.f};
    f32x8 acc[4][2];
#pragma unroll
    for (int i = 0; i < 4; ++i)
#pragma unroll
        for (int j = 0; j < 2; ++j) acc[i][j] = vzero;

    const int ktiles = K >> 5;

#if USE_ASYNC_LDS
    // ---- prologue: async-copy tile 0 into buffer 0 ----
    async_ld128(Ag + (size_t)(m0 + row0) * lda + cc8,      &sA[0][soff]);
    async_ld128(Ag + (size_t)(m0 + row0 + 64) * lda + cc8, &sA[0][soff + 64 * 32]);
    async_ld128(Bg + (size_t)(n0 + row0) * ldb + cc8,      &sB[0][soff]);
    async_ld128(Bg + (size_t)(n0 + row0 + 64) * ldb + cc8, &sB[0][soff + 64 * 32]);

    for (int kt = 0; kt < ktiles; ++kt) {
        const int buf = kt & 1;
        __builtin_amdgcn_s_wait_asynccnt(0);   // this wave's copies done
        __syncthreads();                       // all waves' copies visible;
                                               // also: buf^1 reads finished
        // ---- pull fragments into VGPRs ----
        bf16x16 af[4], bfr[2];
#pragma unroll
        for (int mf = 0; mf < 4; ++mf) {
            int row = wm * 64 + mf * 16 + l15;
            bf16x8 lo = *reinterpret_cast<const bf16x8*>(&sA[buf][row * 32 + khalf * 16]);
            bf16x8 hi = *reinterpret_cast<const bf16x8*>(&sA[buf][row * 32 + 8 + khalf * 16]);
            af[mf] = __builtin_shufflevector(lo, hi, 0, 1, 2, 3, 4, 5, 6, 7,
                                             8, 9, 10, 11, 12, 13, 14, 15);
        }
#pragma unroll
        for (int nf = 0; nf < 2; ++nf) {
            int col = wn * 32 + nf * 16 + l15;
            bfr[nf] = *reinterpret_cast<const bf16x16*>(&sB[buf][col * 32 + khalf * 16]);
        }
        // ---- overlap: issue next tile's DMA into the other buffer ----
        if (kt + 1 < ktiles) {
            const int k0 = (kt + 1) << 5;
            const int nb = buf ^ 1;
            async_ld128(Ag + (size_t)(m0 + row0) * lda + k0 + cc8,      &sA[nb][soff]);
            async_ld128(Ag + (size_t)(m0 + row0 + 64) * lda + k0 + cc8, &sA[nb][soff + 64 * 32]);
            async_ld128(Bg + (size_t)(n0 + row0) * ldb + k0 + cc8,      &sB[nb][soff]);
            async_ld128(Bg + (size_t)(n0 + row0 + 64) * ldb + k0 + cc8, &sB[nb][soff + 64 * 32]);
        }
        // ---- 8 WMMAs ----
#pragma unroll
        for (int mf = 0; mf < 4; ++mf)
#pragma unroll
            for (int nf = 0; nf < 2; ++nf)
                acc[mf][nf] = __builtin_amdgcn_wmma_f32_16x16x32_bf16(
                    false, af[mf], false, bfr[nf], (short)0, acc[mf][nf],
                    false, false);
    }
#else
    // ---- fallback: register-staged loads, single LDS buffer ----
    for (int kt = 0; kt < ktiles; ++kt) {
        const int k0 = kt << 5;
        uint4 ra0 = *reinterpret_cast<const uint4*>(Ag + (size_t)(m0 + row0) * lda + k0 + cc8);
        uint4 ra1 = *reinterpret_cast<const uint4*>(Ag + (size_t)(m0 + row0 + 64) * lda + k0 + cc8);
        uint4 rb0 = *reinterpret_cast<const uint4*>(Bg + (size_t)(n0 + row0) * ldb + k0 + cc8);
        uint4 rb1 = *reinterpret_cast<const uint4*>(Bg + (size_t)(n0 + row0 + 64) * ldb + k0 + cc8);
        if (kt + 1 < ktiles) {
            __builtin_prefetch(Ag + (size_t)(m0 + (tid >> 1)) * lda + k0 + 32, 0, 1);
            __builtin_prefetch(Bg + (size_t)(n0 + (tid >> 1)) * ldb + k0 + 32, 0, 1);
        }
        __syncthreads();
        *reinterpret_cast<uint4*>(&sA[0][soff]) = ra0;
        *reinterpret_cast<uint4*>(&sA[0][soff + 64 * 32]) = ra1;
        *reinterpret_cast<uint4*>(&sB[0][soff]) = rb0;
        *reinterpret_cast<uint4*>(&sB[0][soff + 64 * 32]) = rb1;
        __syncthreads();

        bf16x16 af[4], bfr[2];
#pragma unroll
        for (int mf = 0; mf < 4; ++mf) {
            int row = wm * 64 + mf * 16 + l15;
            bf16x8 lo = *reinterpret_cast<const bf16x8*>(&sA[0][row * 32 + khalf * 16]);
            bf16x8 hi = *reinterpret_cast<const bf16x8*>(&sA[0][row * 32 + 8 + khalf * 16]);
            af[mf] = __builtin_shufflevector(lo, hi, 0, 1, 2, 3, 4, 5, 6, 7,
                                             8, 9, 10, 11, 12, 13, 14, 15);
        }
#pragma unroll
        for (int nf = 0; nf < 2; ++nf) {
            int col = wn * 32 + nf * 16 + l15;
            bfr[nf] = *reinterpret_cast<const bf16x16*>(&sB[0][col * 32 + khalf * 16]);
        }
#pragma unroll
        for (int mf = 0; mf < 4; ++mf)
#pragma unroll
            for (int nf = 0; nf < 2; ++nf)
                acc[mf][nf] = __builtin_amdgcn_wmma_f32_16x16x32_bf16(
                    false, af[mf], false, bfr[nf], (short)0, acc[mf][nf],
                    false, false);
    }
#endif

    // ---- epilogue (lane0-15: N=lane, M=r ; lane16-31: N=lane-16, M=8+r) ----
#pragma unroll
    for (int mf = 0; mf < 4; ++mf)
#pragma unroll
        for (int nf = 0; nf < 2; ++nf) {
            int colg  = n0 + wn * 32 + nf * 16 + l15;
            int rbase = m0 + wm * 64 + mf * 16 + (khalf << 3);
#pragma unroll
            for (int r = 0; r < 8; ++r) {
                int rowg = rbase + r;
                size_t i = (size_t)rowg * ldc + colg;
                float  v = acc[mf][nf][r];
                if      (EPI == 0) Cg[i] = v;
                else if (EPI == 1) Cg[i] = v + Xg[i];
                else if (EPI == 2) Cg[i] += Xg[(size_t)rowg * rss] * v;
                else               Cg[i] = Xg[(size_t)rowg * rss] * v;
            }
        }
}

// ---------------------------------------------------------------------------
// Tiled transpose + f32->bf16 convert: in[R][C] -> out[C][R]
// ---------------------------------------------------------------------------
__global__ __launch_bounds__(256)
void k_tcvt_bf16(const float* __restrict__ in, bf16* __restrict__ out,
                 int R, int C, long strIn, long strOut)
{
    __shared__ float tile[32][33];
    in  += (long)blockIdx.z * strIn;
    out += (long)blockIdx.z * strOut;
    const int r0 = blockIdx.y * 32;
    const int c0 = blockIdx.x * 32;
    const int tx = threadIdx.x & 31;
    const int ty = threadIdx.x >> 5;        // 0..7
#pragma unroll
    for (int i = ty; i < 32; i += 8)
        tile[i][tx] = in[(size_t)(r0 + i) * C + c0 + tx];
    __syncthreads();
#pragma unroll
    for (int i = ty; i < 32; i += 8)
        out[(size_t)(c0 + i) * R + r0 + tx] = (bf16)tile[tx][i];
}

// ---------------------------------------------------------------------------
// Elementwise / reduction helper kernels
// ---------------------------------------------------------------------------
__global__ __launch_bounds__(256)
void k_rmsnorm_bf16(const float* __restrict__ x, const float* __restrict__ w,
                    bf16* __restrict__ out)
{
    __shared__ float red[256];
    const int t = blockIdx.x;
    const float* xr = x + (size_t)t * ND;
    float ss = 0.f;
    for (int d = threadIdx.x; d < ND; d += 256) { float v = xr[d]; ss += v * v; }
    red[threadIdx.x] = ss; __syncthreads();
    for (int s = 128; s > 0; s >>= 1) {
        if (threadIdx.x < s) red[threadIdx.x] += red[threadIdx.x + s];
        __syncthreads();
    }
    float rs = rsqrtf(red[0] / (float)ND + EPSV);
    for (int d = threadIdx.x; d < ND; d += 256)
        out[(size_t)t * ND + d] = (bf16)(xr[d] * rs * (1.f + w[d]));
}

__global__ void k_rope_bf16(const float* __restrict__ q,
                            const float* __restrict__ cs,
                            const float* __restrict__ sn,
                            bf16* __restrict__ out, size_t n)
{
    size_t i = (size_t)blockIdx.x * blockDim.x + threadIdx.x;
    size_t st = (size_t)gridDim.x * blockDim.x;
    for (; i < n; i += st) {
        int    d = (int)(i & 127);       // DH = 128
        size_t s = i >> 11;              // / (H*DH) = 2048
        float rot = (d < 64) ? -q[i + 64] : q[i - 64];
        out[i] = (bf16)(q[i] * cs[s * 128 + d] + rot * sn[s * 128 + d]);
    }
}

__global__ __launch_bounds__(256)
void k_softmax_bf16(const float* __restrict__ scores,
                    const float* __restrict__ bias,
                    bf16* __restrict__ probs, float scale)
{
    __shared__ float red[256];
    const size_t r    = blockIdx.x;                // H*S rows
    const size_t qrow = r & (size_t)(NS - 1);
    const float* sr = scores + r * (size_t)NS;
    const float* br = bias + qrow * (size_t)NS;
    float mx = -3.0e38f;
    for (int c = threadIdx.x; c < NS; c += 256)
        mx = fmaxf(mx, sr[c] * scale + br[c]);
    red[threadIdx.x] = mx; __syncthreads();
    for (int s = 128; s > 0; s >>= 1) {
        if (threadIdx.x < s) red[threadIdx.x] = fmaxf(red[threadIdx.x], red[threadIdx.x + s]);
        __syncthreads();
    }
    mx = red[0]; __syncthreads();
    float sum = 0.f;
    for (int c = threadIdx.x; c < NS; c += 256)
        sum += expf(sr[c] * scale + br[c] - mx);
    red[threadIdx.x] = sum; __syncthreads();
    for (int s = 128; s > 0; s >>= 1) {
        if (threadIdx.x < s) red[threadIdx.x] += red[threadIdx.x + s];
        __syncthreads();
    }
    float inv = 1.f / red[0];
    for (int c = threadIdx.x; c < NS; c += 256)
        probs[r * (size_t)NS + c] = (bf16)(expf(sr[c] * scale + br[c] - mx) * inv);
}

__global__ void k_sigmul_bf16(const float* __restrict__ a,
                              const float* __restrict__ g,
                              bf16* __restrict__ out, size_t n)
{
    size_t i = (size_t)blockIdx.x * blockDim.x + threadIdx.x;
    size_t st = (size_t)gridDim.x * blockDim.x;
    for (; i < n; i += st)
        out[i] = (bf16)(a[i] * (1.f / (1.f + expf(-g[i]))));
}

__global__ void k_silu_mul_bf16(const float* __restrict__ g,
                                const float* __restrict__ u,
                                bf16* __restrict__ out, size_t n)
{
    size_t i = (size_t)blockIdx.x * blockDim.x + threadIdx.x;
    size_t st = (size_t)gridDim.x * blockDim.x;
    for (; i < n; i += st)
        out[i] = (bf16)((g[i] / (1.f + expf(-g[i]))) * u[i]);
}

__global__ __launch_bounds__(256)
void k_router(const bf16* __restrict__ y, const float* __restrict__ rw,
              const float* __restrict__ sgw, float* __restrict__ wdense,
              float* __restrict__ shsig)
{
    __shared__ float red[256];
    __shared__ float res[9];
    const int t = blockIdx.x;
    const bf16* yr = y + (size_t)t * ND;
    float acc[9];
#pragma unroll
    for (int e = 0; e < 9; ++e) acc[e] = 0.f;
    for (int d = threadIdx.x; d < ND; d += 256) {
        float yv = (float)yr[d];
#pragma unroll
        for (int e = 0; e < 8; ++e) acc[e] += yv * rw[(size_t)d * 8 + e];
        acc[8] += yv * sgw[d];
    }
    for (int e = 0; e < 9; ++e) {
        red[threadIdx.x] = acc[e]; __syncthreads();
        for (int s = 128; s > 0; s >>= 1) {
            if (threadIdx.x < s) red[threadIdx.x] += red[threadIdx.x + s];
            __syncthreads();
        }
        if (threadIdx.x == 0) res[e] = red[0];
        __syncthreads();
    }
    if (threadIdx.x == 0) {
        float p[8], mx = res[0];
        for (int e = 1; e < 8; ++e) mx = fmaxf(mx, res[e]);
        float sum = 0.f;
        for (int e = 0; e < 8; ++e) { p[e] = expf(res[e] - mx); sum += p[e]; }
        for (int e = 0; e < 8; ++e) p[e] /= sum;
        int i0 = 0;
        for (int e = 1; e < 8; ++e) if (p[e] > p[i0]) i0 = e;
        int i1 = (i0 == 0) ? 1 : 0;
        for (int e = 0; e < 8; ++e) if (e != i0 && p[e] > p[i1]) i1 = e;
        float inv = 1.f / (p[i0] + p[i1]);
        for (int e = 0; e < 8; ++e) wdense[(size_t)t * 8 + e] = 0.f;
        wdense[(size_t)t * 8 + i0] = p[i0] * inv;
        wdense[(size_t)t * 8 + i1] = p[i1] * inv;
        shsig[t] = 1.f / (1.f + expf(-res[8]));
    }
}

__global__ void k_add(const float* __restrict__ a, const float* __restrict__ b,
                      float* __restrict__ o, size_t n)
{
    size_t i = (size_t)blockIdx.x * blockDim.x + threadIdx.x;
    size_t st = (size_t)gridDim.x * blockDim.x;
    for (; i < n; i += st) o[i] = a[i] + b[i];
}

// ---------------------------------------------------------------------------
extern "C" void kernel_launch(void* const* d_in, const int* in_sizes, int n_in,
                              void* d_out, int out_size, void* d_ws, size_t ws_size,
                              hipStream_t stream)
{
    const float* hidden = (const float*)d_in[0];
    const float* bias   = (const float*)d_in[1];
    const float* cosp   = (const float*)d_in[2];
    const float* sinp   = (const float*)d_in[3];
    const float* ln1    = (const float*)d_in[4];
    const float* ln2    = (const float*)d_in[5];
    const float* wq     = (const float*)d_in[6];
    const float* wk     = (const float*)d_in[7];
    const float* wv     = (const float*)d_in[8];
    const float* wg     = (const float*)d_in[9];
    const float* wo     = (const float*)d_in[10];
    const float* rw     = (const float*)d_in[11];
    const float* eg     = (const float*)d_in[12];
    const float* eu     = (const float*)d_in[13];
    const float* ed     = (const float*)d_in[14];
    const float* shg    = (const float*)d_in[15];
    const float* shu    = (const float*)d_in[16];
    const float* shd    = (const float*)d_in[17];
    const float* sgw    = (const float*)d_in[18];
    float* out = (float*)d_out;

    const size_t SD  = (size_t)NS * ND;
    const size_t SS  = (size_t)NS * NS;
    const size_t EDF = (size_t)NE * ND * NF;
    const size_t ESF = (size_t)NE * NS * NF;
    const size_t DSF = (size_t)ND * NSF;
    const size_t SSF = (size_t)NS * NSF;

    char* p = (char*)d_ws;
    auto alloc = [&](size_t bytes) -> char* {
        char* r = p; p += (bytes + 255) & ~(size_t)255; return r;
    };

    // bf16 buffers (all *_t weights stored transposed: [N][K])
    bf16* wq_t   = (bf16*)alloc(SD * 2);
    bf16* wk_t   = (bf16*)alloc(SD * 2);
    bf16* wv_t   = (bf16*)alloc(SD * 2);
    bf16* wg_t   = (bf16*)alloc(SD * 2);
    bf16* wo_t   = (bf16*)alloc(SD * 2);
    bf16* eg_t   = (bf16*)alloc(EDF * 2);
    bf16* eu_t   = (bf16*)alloc(EDF * 2);
    bf16* ed_t   = (bf16*)alloc(EDF * 2);
    bf16* shg_t  = (bf16*)alloc(DSF * 2);
    bf16* shu_t  = (bf16*)alloc(DSF * 2);
    bf16* shd_t  = (bf16*)alloc(DSF * 2);
    bf16* xn_bf  = (bf16*)alloc(SD * 2);
    bf16* q_bf   = (bf16*)alloc(SD * 2);
    bf16* k_bf   = (bf16*)alloc(SD * 2);
    bf16* vt     = (bf16*)alloc(SD * 2);        // v transposed [HD][S]
    bf16* probs  = (bf16*)alloc((size_t)NH * SS * 2);
    bf16* ctx_bf = (bf16*)alloc(SD * 2);
    bf16* y_bf   = (bf16*)alloc(SD * 2);
    bf16* shact  = (bf16*)alloc(SSF * 2);
    bf16* exact  = (bf16*)alloc(ESF * 2);
    // f32 buffers
    float* qf    = (float*)alloc(SD * 4);
    float* kf    = (float*)alloc(SD * 4);
    float* vf    = (float*)alloc(SD * 4);
    float* gaf   = (float*)alloc(SD * 4);
    float* scrs  = (float*)alloc((size_t)NH * SS * 4);
    float* ctxf  = (float*)alloc(SD * 4);
    float* hbuf  = (float*)alloc(SD * 4);
    float* moe   = (float*)alloc(SD * 4);
    float* shgf  = (float*)alloc(SSF * 4);
    float* shuf  = (float*)alloc(SSF * 4);
    float* exgf  = (float*)alloc(ESF * 4);
    float* exuf  = (float*)alloc(ESF * 4);
    float* wden  = (float*)alloc((size_t)NS * NE * 4);
    float* shsg  = (float*)alloc((size_t)NS * 4);

    const dim3 B256(256);
    const dim3 GEW(4096);
    const float ATT_SCALE = 0.08838834764831845f;  // 1/sqrt(128)

    // 1) weights -> bf16, transposed to [N][K]
    k_tcvt_bf16<<<dim3(64, 64, 1),  B256, 0, stream>>>(wq,  wq_t,  ND,  NHD, 0, 0);
    k_tcvt_bf16<<<dim3(64, 64, 1),  B256, 0, stream>>>(wk,  wk_t,  ND,  NHD, 0, 0);
    k_tcvt_bf16<<<dim3(64, 64, 1),  B256, 0, stream>>>(wv,  wv_t,  ND,  NHD, 0, 0);
    k_tcvt_bf16<<<dim3(64, 64, 1),  B256, 0, stream>>>(wg,  wg_t,  ND,  NHD, 0, 0);
    k_tcvt_bf16<<<dim3(64, 64, 1),  B256, 0, stream>>>(wo,  wo_t,  NHD, ND,  0, 0);
    k_tcvt_bf16<<<dim3(24, 64, NE), B256, 0, stream>>>(eg,  eg_t,  ND,  NF,
                                                       (long)ND * NF, (long)ND * NF);
    k_tcvt_bf16<<<dim3(24, 64, NE), B256, 0, stream>>>(eu,  eu_t,  ND,  NF,
                                                       (long)ND * NF, (long)ND * NF);
    k_tcvt_bf16<<<dim3(64, 24, NE), B256, 0, stream>>>(ed,  ed_t,  NF,  ND,
                                                       (long)NF * ND, (long)NF * ND);
    k_tcvt_bf16<<<dim3(128, 64, 1), B256, 0, stream>>>(shg, shg_t, ND,  NSF, 0, 0);
    k_tcvt_bf16<<<dim3(128, 64, 1), B256, 0, stream>>>(shu, shu_t, ND,  NSF, 0, 0);
    k_tcvt_bf16<<<dim3(64, 128, 1), B256, 0, stream>>>(shd, shd_t, NSF, ND,  0, 0);

    // 2) ln1 -> xn_bf
    k_rmsnorm_bf16<<<NS, B256, 0, stream>>>(hidden, ln1, xn_bf);

    // 3) q/k/v/gate projections (2048^3 each)
    const dim3 Gqkv(ND / 128, NS / 128, 1);
    k_gemm_wmma<0><<<Gqkv, B256, 0, stream>>>(xn_bf, wq_t, qf,  nullptr,
        ND, ND, ND, ND, 0, 0, 0, 0);
    k_gemm_wmma<0><<<Gqkv, B256, 0, stream>>>(xn_bf, wk_t, kf,  nullptr,
        ND, ND, ND, ND, 0, 0, 0, 0);
    k_gemm_wmma<0><<<Gqkv, B256, 0, stream>>>(xn_bf, wv_t, vf,  nullptr,
        ND, ND, ND, ND, 0, 0, 0, 0);
    k_gemm_wmma<0><<<Gqkv, B256, 0, stream>>>(xn_bf, wg_t, gaf, nullptr,
        ND, ND, ND, ND, 0, 0, 0, 0);

    // 4) RoPE (q,k) -> bf16; v -> bf16 transposed per head ([HD][S])
    k_rope_bf16<<<GEW, B256, 0, stream>>>(qf, cosp, sinp, q_bf, SD);
    k_rope_bf16<<<GEW, B256, 0, stream>>>(kf, cosp, sinp, k_bf, SD);
    k_tcvt_bf16<<<dim3(64, 64, 1), B256, 0, stream>>>(vf, vt, NS, NHD, 0, 0);

    // 5) scores = q @ k^T per head (k is naturally [N][K]; z=16)
    k_gemm_wmma<0><<<dim3(NS / 128, NS / 128, NH), B256, 0, stream>>>(
        q_bf, k_bf, scrs, nullptr, NDH, NHD, NHD, NS,
        (long)NDH, (long)NDH, (long)SS, 0);

    // 6) softmax rows (scale + bias)
    k_softmax_bf16<<<NH * NS, B256, 0, stream>>>(scrs, bias, probs, ATT_SCALE);

    // 7) ctx = probs @ v per head (B = vt[h*128.., :], z=16)
    k_gemm_wmma<0><<<dim3(1, NS / 128, NH), B256, 0, stream>>>(
        probs, vt, ctxf, nullptr, NS, NS, NS, NHD,
        (long)SS, (long)NDH * NS, (long)NDH, 0);

    // 8) gate: ctx_bf = bf16(ctx * sigmoid(gaf))
    k_sigmul_bf16<<<GEW, B256, 0, stream>>>(ctxf, gaf, ctx_bf, SD);

    // 9) h = hidden + ctx_bf @ wo
    k_gemm_wmma<1><<<Gqkv, B256, 0, stream>>>(ctx_bf, wo_t, hbuf, hidden,
        ND, ND, ND, ND, 0, 0, 0, 0);

    // 10) ln2 -> y_bf
    k_rmsnorm_bf16<<<NS, B256, 0, stream>>>(hbuf, ln2, y_bf);

    // 11) router top-2 weights + shared-expert gate scalar
    k_router<<<NS, B256, 0, stream>>>(y_bf, rw, sgw, wden, shsg);

    // 12) shared expert (seeds moe with EPI=3 store-scaled)
    const dim3 Gsh(NSF / 128, NS / 128, 1);
    k_gemm_wmma<0><<<Gsh, B256, 0, stream>>>(y_bf, shg_t, shgf, nullptr,
        ND, ND, ND, NSF, 0, 0, 0, 0);
    k_gemm_wmma<0><<<Gsh, B256, 0, stream>>>(y_bf, shu_t, shuf, nullptr,
        ND, ND, ND, NSF, 0, 0, 0, 0);
    k_silu_mul_bf16<<<GEW, B256, 0, stream>>>(shgf, shuf, shact, SSF);
    k_gemm_wmma<3><<<Gqkv, B256, 0, stream>>>(shact, shd_t, moe, shsg,
        NSF, NSF, NSF, ND, 0, 0, 0, 1);

    // 13) experts: batched gate/up (z=8), then 8 sequential down-GEMMs
    //     accumulating with per-row scale w_dense[:,e] (deterministic)
    const dim3 Gexp(NF / 128, NS / 128, NE);
    k_gemm_wmma<0><<<Gexp, B256, 0, stream>>>(y_bf, eg_t, exgf, nullptr,
        ND, ND, ND, NF, 0, (long)ND * NF, (long)NS * NF, 0);
    k_gemm_wmma<0><<<Gexp, B256, 0, stream>>>(y_bf, eu_t, exuf, nullptr,
        ND, ND, ND, NF, 0, (long)ND * NF, (long)NS * NF, 0);
    k_silu_mul_bf16<<<GEW, B256, 0, stream>>>(exgf, exuf, exact, ESF);
    for (int e = 0; e < NE; ++e) {
        k_gemm_wmma<2><<<Gqkv, B256, 0, stream>>>(
            exact + (size_t)e * NS * NF, ed_t + (size_t)e * NF * ND, moe,
            wden + e, NF, NF, NF, ND, 0, 0, 0, NE);
    }

    // 14) out = h + moe
    k_add<<<GEW, B256, 0, stream>>>(hbuf, moe, out, SD);
}
